// CapsuleNetwork_48928267436070
// MI455X (gfx1250) — matrix operand order repaired and checked by reading, not compiled, not run on hardware
//
// CapsuleNetwork (BiLSTM x2 + attention + capsule routing) for MI455X / gfx1250.
// All GEMM-shaped math on v_wmma_f32_16x16x32_bf16 (wave32, f32 accumulate).
// ~340 GFLOP total vs ~1.5 GB traffic -> compute/latency shaped on MI455X.
// Critical path = 128 sequential LSTM steps: fused WMMA-GEMM + gate kernel per
// step, gate-interleaved xW layout so the epilogue is one b128 load per elem.
#include <hip/hip_runtime.h>

#define B_   256
#define T_   64
#define E_   300
#define EP   320      // E padded to K-multiple of 32
#define H_   512
#define H4   2048     // 4*H
#define H2   1024     // 2*H
#define DA_  350
#define DAP  384      // DA padded to N-tile multiple of 128
#define R_   16
#define S_   64
#define A_   16
#define NR_  3
#define MR   (T_*B_)  // 16384 rows for time-parallel GEMMs

typedef __attribute__((ext_vector_type(8)))  __bf16 bf16x8;
typedef __attribute__((ext_vector_type(16))) __bf16 bf16x16;
typedef __attribute__((ext_vector_type(8)))  float  floatx8;

// ---------------------------------------------------------------- fragments
// A operand: 16x32 bf16. lanes 0-15: M=lane, K = {0..7}U{16..23};
// lanes 16-31: M=lane-16, K = {8..15}U{24..31}.  (ISA 7.12.2)
__device__ __forceinline__ bf16x16 load_a_frag(const __bf16* __restrict__ base,
                                               int ld, int m0, int k0) {
  int lane = threadIdx.x & 31;
  int half = lane >> 4;
  int r    = lane & 15;
  const __bf16* p = base + (size_t)(m0 + r) * ld + k0 + 8 * half;
  bf16x8 lo = *(const bf16x8*)p;
  bf16x8 hi = *(const bf16x8*)(p + 16);
  bf16x16 f;
#pragma unroll
  for (int i = 0; i < 8; ++i) { f[i] = lo[i]; f[i + 8] = hi[i]; }
  return f;
}

// B operand: 32x16 bf16 = W^T where W is (N,K) row-major. lanes 0-15: N=lane,
// K = {0..7}U{16..23}; lanes 16-31: same N, K = {8..15}U{24..31}.
__device__ __forceinline__ bf16x16 load_b_frag(const __bf16* __restrict__ base,
                                               int ld, int n0, int k0) {
  int lane = threadIdx.x & 31;
  int half = lane >> 4;
  int n    = lane & 15;
  const __bf16* p = base + (size_t)(n0 + n) * ld + k0 + 8 * half;
  bf16x8 lo = *(const bf16x8*)p;
  bf16x8 hi = *(const bf16x8*)(p + 16);
  bf16x16 f;
#pragma unroll
  for (int i = 0; i < 8; ++i) { f[i] = lo[i]; f[i + 8] = hi[i]; }
  return f;
}

__device__ __forceinline__ floatx8 wmma_bf16(bf16x16 a, bf16x16 b, floatx8 c) {
  return __builtin_amdgcn_wmma_f32_16x16x32_bf16(false, a, false, b,
                                                 (short)0, c, false, false);
}

__device__ __forceinline__ float sigm(float x) {
  return 1.0f / (1.0f + __expf(-x));
}

// ---------------------------------------------------------------- utilities
__global__ void zero_u32(unsigned int* p, long n) {
  long i = (long)blockIdx.x * blockDim.x + threadIdx.x;
  if (i < n) p[i] = 0u;
}

// f32 (srows,scols) -> bf16 (drows,dcols), zero padded.
__global__ void cvt_pad_bf16(const float* __restrict__ src, __bf16* __restrict__ dst,
                             int srows, int scols, int drows, int dcols) {
  long idx = (long)blockIdx.x * blockDim.x + threadIdx.x;
  long tot = (long)drows * dcols;
  if (idx >= tot) return;
  int r = (int)(idx / dcols), c = (int)(idx % dcols);
  float v = (r < srows && c < scols) ? src[(long)r * scols + c] : 0.0f;
  dst[idx] = (__bf16)v;
}

// caps_W (R, I=1024, O=1024) f32 -> (R, O, I) bf16 so it is a (N,K) B-operand.
__global__ void cvt_caps(const float* __restrict__ src, __bf16* __restrict__ dst) {
  long idx = (long)blockIdx.x * blockDim.x + threadIdx.x;
  if (idx >= (long)R_ * H2 * H2) return;
  int o = (int)(idx & 1023);
  int i = (int)((idx >> 10) & 1023);
  int r = (int)(idx >> 20);
  dst[((long)r << 20) | ((long)o << 10) | i] = (__bf16)src[idx];
}

// tokens (B,T) -> x bf16 (T*B, EP) zero padded, time-major rows m = t*B+b.
__global__ void embed_gather(const int* __restrict__ tokens,
                             const float* __restrict__ emb,
                             __bf16* __restrict__ x) {
  int row = blockIdx.x;                 // t*B + b
  int t = row / B_, b = row % B_;
  int tok = tokens[b * T_ + t];
  const float* e = emb + (long)tok * E_;
  for (int c = threadIdx.x; c < EP; c += blockDim.x)
    x[(long)row * EP + c] = (c < E_) ? (__bf16)e[c] : (__bf16)0.0f;
}

// ------------------------------------------------- 128x128 WMMA GEMM (bf16)
// C[m,n] = act( sum_k A[m,k]*W[n,k] + bias[n] ).  8 waves, each 64x32 via
// 4x2 wmma tiles.  M%128==0, N%128==0, K%32==0.  Template flags keep the
// epilogue branch-free.  GILV stores C gate-interleaved for the LSTM:
// n -> (n&511)*4 + (n>>9), so the recurrent kernel reads one float4/elem.
template <bool BIAS, bool TANH, bool OUTF, bool OUTB, bool GILV>
__global__ __launch_bounds__(256) void gemm128(
    const __bf16* __restrict__ A, const __bf16* __restrict__ W,
    const float* __restrict__ bias, float* __restrict__ C,
    __bf16* __restrict__ Cbf, int K, int lda, int ldw, int ldc,
    long sA, long sW, long sC) {
  A += (long)blockIdx.z * sA;
  W += (long)blockIdx.z * sW;
  long cOff = (long)blockIdx.z * sC;
  int lane = threadIdx.x & 31, wid = threadIdx.x >> 5;
  int m0 = blockIdx.y * 128 + (wid >> 2) * 64;
  int n0 = blockIdx.x * 128 + (wid & 3) * 32;

  floatx8 acc[4][2];
#pragma unroll
  for (int i = 0; i < 4; ++i)
#pragma unroll
    for (int j = 0; j < 2; ++j) acc[i][j] = (floatx8)0.0f;

  for (int k0 = 0; k0 < K; k0 += 32) {
    if (k0 + 32 < K) {  // warm WGP$ for the streamed weight rows (L2-resident)
      __builtin_prefetch(W + (size_t)(n0 + (lane & 15)) * ldw + k0 + 32, 0, 1);
      __builtin_prefetch(W + (size_t)(n0 + 16 + (lane & 15)) * ldw + k0 + 32, 0, 1);
    }
    bf16x16 a[4], w[2];
#pragma unroll
    for (int i = 0; i < 4; ++i) a[i] = load_a_frag(A, lda, m0 + 16 * i, k0);
#pragma unroll
    for (int j = 0; j < 2; ++j) w[j] = load_b_frag(W, ldw, n0 + 16 * j, k0);
#pragma unroll
    for (int i = 0; i < 4; ++i)
#pragma unroll
      for (int j = 0; j < 2; ++j) acc[i][j] = wmma_bf16(a[i], w[j], acc[i][j]);
  }

  int half = lane >> 4, nc = lane & 15;
  float bv[2] = {0.0f, 0.0f};
  if (BIAS) {  // one load per n-column, not per element
#pragma unroll
    for (int j = 0; j < 2; ++j) bv[j] = bias[n0 + j * 16 + nc];
  }
#pragma unroll
  for (int i = 0; i < 4; ++i)
#pragma unroll
    for (int j = 0; j < 2; ++j) {
      int n = n0 + j * 16 + nc;
      long nidx = GILV ? (long)((n & 511) * 4 + (n >> 9)) : (long)n;
#pragma unroll
      for (int v = 0; v < 8; ++v) {
        int m = m0 + i * 16 + v + 8 * half;
        float val = acc[i][j][v];
        if (BIAS) val += bv[j];
        if (TANH) val = tanhf(val);
        long o = cOff + (long)m * ldc + nidx;
        if (OUTF) C[o] = val;
        if (OUTB) Cbf[o] = (__bf16)val;
      }
    }
}

// Small-N GEMM (N==16) for alphas: 8 waves, one 16x16 tile each.  M%128==0.
__global__ __launch_bounds__(256) void gemm_bf16_Mx16(
    const __bf16* __restrict__ A, const __bf16* __restrict__ W,
    float* __restrict__ C, int K, int lda, int ldw, int ldc) {
  int lane = threadIdx.x & 31, wid = threadIdx.x >> 5;
  int m0 = blockIdx.x * 128 + wid * 16;
  floatx8 acc = (floatx8)0.0f;
  for (int k0 = 0; k0 < K; k0 += 32) {
    bf16x16 a = load_a_frag(A, lda, m0, k0);
    bf16x16 w = load_b_frag(W, ldw, 0, k0);
    acc = wmma_bf16(a, w, acc);
  }
  int half = lane >> 4, nc = lane & 15;
#pragma unroll
  for (int v = 0; v < 8; ++v)
    C[(long)(m0 + v + 8 * half) * ldc + nc] = acc[v];
}

// ----------------------------------------------------- fused LSTM time step
// z = h_prev @ Whh^T + xw[t]  (xw = x@Wih^T + b precomputed, GATE-INTERLEAVED
// as (row, j, gate) so all 4 gate pre-activations arrive in one b128 load).
// Each block: 64 batch rows x 64 hidden units, all four gate slices, so the
// nonlinearity + (c,h) update is in-register.  8 waves: 32 rows x 16 units x
// 4 gates = 8 accumulators/wave, K=512 -> 128 wmma per wave.
__global__ __launch_bounds__(256) void lstm_step(
    const __bf16* __restrict__ hprevF, const __bf16* __restrict__ hprevB,
    __bf16* __restrict__ hnextF, __bf16* __restrict__ hnextB,
    float* __restrict__ cF, float* __restrict__ cB,
    const __bf16* __restrict__ whhF, const __bf16* __restrict__ whhB,
    const float* __restrict__ xwF, const float* __restrict__ xwB,
    float* __restrict__ seqF32, __bf16* __restrict__ seqBf, int t) {
  int dir = blockIdx.z;
  const __bf16* hprev = dir ? hprevB : hprevF;
  __bf16* hnext = dir ? hnextB : hnextF;
  float* cbuf = dir ? cB : cF;
  const __bf16* whh = dir ? whhB : whhF;
  const float* xw = dir ? xwB : xwF;
  int teff = dir ? (T_ - 1 - t) : t;

  int lane = threadIdx.x & 31, wid = threadIdx.x >> 5;
  int r0 = blockIdx.y * 64 + (wid >> 2) * 32;  // batch rows
  int j0 = blockIdx.x * 64 + (wid & 3) * 16;   // hidden units

  floatx8 acc[4][2];
#pragma unroll
  for (int g = 0; g < 4; ++g)
#pragma unroll
    for (int m = 0; m < 2; ++m) acc[g][m] = (floatx8)0.0f;

  for (int k0 = 0; k0 < H_; k0 += 32) {
    if (k0 + 32 < H_) {
#pragma unroll
      for (int g = 0; g < 4; ++g)
        __builtin_prefetch(whh + (size_t)(g * H_ + j0 + (lane & 15)) * H_ + k0 + 32, 0, 1);
    }
    bf16x16 a0 = load_a_frag(hprev, H_, r0, k0);
    bf16x16 a1 = load_a_frag(hprev, H_, r0 + 16, k0);
#pragma unroll
    for (int g = 0; g < 4; ++g) {
      bf16x16 w = load_b_frag(whh, H_, g * H_ + j0, k0);
      acc[g][0] = wmma_bf16(a0, w, acc[g][0]);
      acc[g][1] = wmma_bf16(a1, w, acc[g][1]);
    }
  }

  int half = lane >> 4;
  int j = j0 + (lane & 15);
#pragma unroll
  for (int mt = 0; mt < 2; ++mt)
#pragma unroll
    for (int v = 0; v < 8; ++v) {
      int row = r0 + mt * 16 + v + 8 * half;
      long xrow = ((long)teff * B_ + row) * H4;
      float4 z4 = *(const float4*)(xw + xrow + 4 * j);  // gates i,f,g,o
      float zi = acc[0][mt][v] + z4.x;
      float zf = acc[1][mt][v] + z4.y;
      float zg = acc[2][mt][v] + z4.z;
      float zo = acc[3][mt][v] + z4.w;
      float cp = cbuf[row * H_ + j];
      float cn = sigm(zf) * cp + sigm(zi) * tanhf(zg);
      float h  = sigm(zo) * tanhf(cn);
      cbuf[row * H_ + j] = cn;
      hnext[row * H_ + j] = (__bf16)h;
      long srow = ((long)teff * B_ + row) * H2 + (long)dir * H_ + j;
      seqBf[srow] = (__bf16)h;
      if (seqF32) seqF32[srow] = h;
    }
}

// ----------------------------------------------------- sent = alphas^T @ l1
// sent[b,r,e] = sum_t alphas[t*B+b, r] * l1[t*B+b, e].  268 M MAC -> VALU ok.
__global__ __launch_bounds__(256) void sent_kernel(
    const float* __restrict__ alphas, const float* __restrict__ l1f,
    float* __restrict__ outSent, __bf16* __restrict__ sentBf) {
  int b = blockIdx.y;
  int e = blockIdx.x * 256 + threadIdx.x;
  __shared__ float al[T_ * R_];
  for (int idx = threadIdx.x; idx < T_ * R_; idx += 256) {
    int tt = idx >> 4, r = idx & 15;
    al[idx] = alphas[((long)tt * B_ + b) * R_ + r];
  }
  __syncthreads();
  float acc[R_];
#pragma unroll
  for (int r = 0; r < R_; ++r) acc[r] = 0.0f;
  for (int tt = 0; tt < T_; ++tt) {
    float x = l1f[((long)tt * B_ + b) * H2 + e];
#pragma unroll
    for (int r = 0; r < R_; ++r) acc[r] = fmaf(al[tt * R_ + r], x, acc[r]);
  }
#pragma unroll
  for (int r = 0; r < R_; ++r) {
    long o = ((long)b * R_ + r) * H2 + e;
    outSent[o] = acc[r];
    sentBf[o] = (__bf16)acc[r];
  }
}

// ------------------------------------------------------- dynamic routing
__global__ __launch_bounds__(256) void routing_kernel(
    const float* __restrict__ votes, float* __restrict__ classLogits) {
  int b = blockIdx.x;
  const float* vb = votes + (long)b * R_ * S_ * A_;
  __shared__ float logits[R_ * S_];
  __shared__ float route[R_ * S_];
  __shared__ float preact[S_ * A_];
  __shared__ float scl[S_];
  int tid = threadIdx.x;
  for (int i = tid; i < R_ * S_; i += 256) logits[i] = 0.0f;
  __syncthreads();
  for (int it = 0; it < NR_; ++it) {
    if (tid < R_) {  // softmax over s for row r = tid
      float mx = -1e30f;
      for (int s = 0; s < S_; ++s) mx = fmaxf(mx, logits[tid * S_ + s]);
      float sum = 0.0f;
      for (int s = 0; s < S_; ++s) {
        float e = __expf(logits[tid * S_ + s] - mx);
        route[tid * S_ + s] = e;
        sum += e;
      }
      float inv = 1.0f / sum;
      for (int s = 0; s < S_; ++s) route[tid * S_ + s] *= inv;
    }
    __syncthreads();
    for (int idx = tid; idx < S_ * A_; idx += 256) {
      int s = idx >> 4, a = idx & 15;
      float p = 0.0f;
      for (int r = 0; r < R_; ++r)
        p = fmaf(route[r * S_ + s], vb[(r * S_ + s) * A_ + a], p);
      preact[idx] = p;
    }
    __syncthreads();
    if (tid < S_) {
      float n2 = 0.0f;
      for (int a = 0; a < A_; ++a) {
        float p = preact[tid * A_ + a];
        n2 = fmaf(p, p, n2);
      }
      scl[tid] = sqrtf(n2) / (0.5f + n2);  // act = preact * norm/(0.5+norm^2)
    }
    __syncthreads();
    for (int idx = tid; idx < R_ * S_; idx += 256) {
      int r = idx / S_, s = idx % S_;
      float d = 0.0f;
      for (int a = 0; a < A_; ++a)
        d = fmaf(vb[(r * S_ + s) * A_ + a], preact[s * A_ + a] * scl[s], d);
      logits[idx] += d;
    }
    __syncthreads();
  }
  if (tid < S_) {
    float n2 = 0.0f;
    for (int a = 0; a < A_; ++a) {
      float ac = preact[tid * A_ + a] * scl[tid];
      n2 = fmaf(ac, ac, n2);
    }
    classLogits[b * S_ + tid] = sqrtf(n2);
  }
}

// ------------------------------------------------------------ orchestration
extern "C" void kernel_launch(void* const* d_in, const int* in_sizes, int n_in,
                              void* d_out, int out_size, void* d_ws, size_t ws_size,
                              hipStream_t stream) {
  (void)in_sizes; (void)n_in; (void)out_size; (void)ws_size;
  const int*   tokens = (const int*)d_in[0];
  const float* emb    = (const float*)d_in[2];
  const float* Wih0f  = (const float*)d_in[3];
  const float* Whh0f  = (const float*)d_in[4];
  const float* b0f    = (const float*)d_in[5];
  const float* Wih0b  = (const float*)d_in[6];
  const float* Whh0b  = (const float*)d_in[7];
  const float* b0b    = (const float*)d_in[8];
  const float* Wih1f  = (const float*)d_in[9];
  const float* Whh1f  = (const float*)d_in[10];
  const float* b1f    = (const float*)d_in[11];
  const float* Wih1b  = (const float*)d_in[12];
  const float* Whh1b  = (const float*)d_in[13];
  const float* b1b    = (const float*)d_in[14];
  const float* ws1    = (const float*)d_in[15];
  const float* ws2    = (const float*)d_in[16];
  const float* capsW  = (const float*)d_in[17];
  float* out = (float*)d_out;

  char* wsb = (char*)d_ws;
  size_t off = 0;
  auto take = [&](size_t bytes) -> void* {
    void* p = wsb + off;
    off += (bytes + 255) & ~(size_t)255;
    return p;
  };
  __bf16* xb     = (__bf16*)take((size_t)MR * EP * 2);
  __bf16* wih0fb = (__bf16*)take((size_t)H4 * EP * 2);
  __bf16* wih0bb = (__bf16*)take((size_t)H4 * EP * 2);
  __bf16* whh0fb = (__bf16*)take((size_t)H4 * H_ * 2);
  __bf16* whh0bb = (__bf16*)take((size_t)H4 * H_ * 2);
  __bf16* whh1fb = (__bf16*)take((size_t)H4 * H_ * 2);
  __bf16* whh1bb = (__bf16*)take((size_t)H4 * H_ * 2);
  __bf16* wih1fb = (__bf16*)take((size_t)H4 * H2 * 2);
  __bf16* wih1bb = (__bf16*)take((size_t)H4 * H2 * 2);
  __bf16* ws1b   = (__bf16*)take((size_t)DAP * H2 * 2);
  __bf16* ws2b   = (__bf16*)take((size_t)R_ * DAP * 2);
  __bf16* capsb  = (__bf16*)take((size_t)R_ * H2 * H2 * 2);
  float*  xwf    = (float*)take((size_t)MR * H4 * 4);
  float*  xwb    = (float*)take((size_t)MR * H4 * 4);
  __bf16* l0seq  = (__bf16*)take((size_t)MR * H2 * 2);
  __bf16* l1seqb = (__bf16*)take((size_t)MR * H2 * 2);
  float*  l1seqf = (float*)take((size_t)MR * H2 * 4);
  __bf16* hF0 = (__bf16*)take((size_t)B_ * H_ * 2);
  __bf16* hF1 = (__bf16*)take((size_t)B_ * H_ * 2);
  __bf16* hB0 = (__bf16*)take((size_t)B_ * H_ * 2);
  __bf16* hB1 = (__bf16*)take((size_t)B_ * H_ * 2);
  float*  cF  = (float*)take((size_t)B_ * H_ * 4);
  float*  cB  = (float*)take((size_t)B_ * H_ * 4);
  __bf16* hbar   = (__bf16*)take((size_t)MR * DAP * 2);
  float*  alphas = (float*)take((size_t)MR * R_ * 4);
  __bf16* sentb  = (__bf16*)take((size_t)B_ * R_ * H2 * 2);
  float*  votes  = (float*)take((size_t)B_ * R_ * S_ * A_ * 4);

  auto grid1 = [](long n) { return dim3((unsigned)((n + 255) / 256)); };

  // 0) weights -> bf16 (padded), embedding gather
  embed_gather<<<dim3(MR), 128, 0, stream>>>(tokens, emb, xb);
  cvt_pad_bf16<<<grid1((long)H4 * EP), 256, 0, stream>>>(Wih0f, wih0fb, H4, E_, H4, EP);
  cvt_pad_bf16<<<grid1((long)H4 * EP), 256, 0, stream>>>(Wih0b, wih0bb, H4, E_, H4, EP);
  cvt_pad_bf16<<<grid1((long)H4 * H_), 256, 0, stream>>>(Whh0f, whh0fb, H4, H_, H4, H_);
  cvt_pad_bf16<<<grid1((long)H4 * H_), 256, 0, stream>>>(Whh0b, whh0bb, H4, H_, H4, H_);
  cvt_pad_bf16<<<grid1((long)H4 * H_), 256, 0, stream>>>(Whh1f, whh1fb, H4, H_, H4, H_);
  cvt_pad_bf16<<<grid1((long)H4 * H_), 256, 0, stream>>>(Whh1b, whh1bb, H4, H_, H4, H_);
  cvt_pad_bf16<<<grid1((long)H4 * H2), 256, 0, stream>>>(Wih1f, wih1fb, H4, H2, H4, H2);
  cvt_pad_bf16<<<grid1((long)H4 * H2), 256, 0, stream>>>(Wih1b, wih1bb, H4, H2, H4, H2);
  cvt_pad_bf16<<<grid1((long)DAP * H2), 256, 0, stream>>>(ws1, ws1b, DA_, H2, DAP, H2);
  cvt_pad_bf16<<<grid1((long)R_ * DAP), 256, 0, stream>>>(ws2, ws2b, R_, DA_, R_, DAP);
  cvt_caps<<<grid1((long)R_ * H2 * H2), 256, 0, stream>>>(capsW, capsb);

  // 1) layer-0 input projections (gate-interleaved): xw = x @ Wih^T + b
  gemm128<true, false, true, false, true>
      <<<dim3(H4 / 128, MR / 128, 1), 256, 0, stream>>>(
      xb, wih0fb, b0f, xwf, nullptr, EP, EP, EP, H4, 0, 0, 0);
  gemm128<true, false, true, false, true>
      <<<dim3(H4 / 128, MR / 128, 1), 256, 0, stream>>>(
      xb, wih0bb, b0b, xwb, nullptr, EP, EP, EP, H4, 0, 0, 0);

  // 2) layer-0 recurrence (fwd+bwd fused per launch, double-buffered h)
  long hstateW = (long)B_ * H_ / 2;  // bf16 elems as u32 words
  zero_u32<<<grid1(hstateW), 256, 0, stream>>>((unsigned int*)hF0, hstateW);
  zero_u32<<<grid1(hstateW), 256, 0, stream>>>((unsigned int*)hB0, hstateW);
  zero_u32<<<grid1((long)B_ * H_), 256, 0, stream>>>((unsigned int*)cF, (long)B_ * H_);
  zero_u32<<<grid1((long)B_ * H_), 256, 0, stream>>>((unsigned int*)cB, (long)B_ * H_);
  for (int t = 0; t < T_; ++t) {
    const __bf16* hpF = (t & 1) ? hF1 : hF0;
    const __bf16* hpB = (t & 1) ? hB1 : hB0;
    __bf16* hnF = (t & 1) ? hF0 : hF1;
    __bf16* hnB = (t & 1) ? hB0 : hB1;
    lstm_step<<<dim3(H_ / 64, B_ / 64, 2), 256, 0, stream>>>(
        hpF, hpB, hnF, hnB, cF, cB, whh0fb, whh0bb, xwf, xwb,
        nullptr, l0seq, t);
  }

  // 3) layer-1 input projections: xw = l0 @ Wih1^T + b (gate-interleaved)
  gemm128<true, false, true, false, true>
      <<<dim3(H4 / 128, MR / 128, 1), 256, 0, stream>>>(
      l0seq, wih1fb, b1f, xwf, nullptr, H2, H2, H2, H4, 0, 0, 0);
  gemm128<true, false, true, false, true>
      <<<dim3(H4 / 128, MR / 128, 1), 256, 0, stream>>>(
      l0seq, wih1bb, b1b, xwb, nullptr, H2, H2, H2, H4, 0, 0, 0);

  // 4) layer-1 recurrence (writes f32 + bf16 sequence for attention)
  zero_u32<<<grid1(hstateW), 256, 0, stream>>>((unsigned int*)hF0, hstateW);
  zero_u32<<<grid1(hstateW), 256, 0, stream>>>((unsigned int*)hB0, hstateW);
  zero_u32<<<grid1((long)B_ * H_), 256, 0, stream>>>((unsigned int*)cF, (long)B_ * H_);
  zero_u32<<<grid1((long)B_ * H_), 256, 0, stream>>>((unsigned int*)cB, (long)B_ * H_);
  for (int t = 0; t < T_; ++t) {
    const __bf16* hpF = (t & 1) ? hF1 : hF0;
    const __bf16* hpB = (t & 1) ? hB1 : hB0;
    __bf16* hnF = (t & 1) ? hF0 : hF1;
    __bf16* hnB = (t & 1) ? hB0 : hB1;
    lstm_step<<<dim3(H_ / 64, B_ / 64, 2), 256, 0, stream>>>(
        hpF, hpB, hnF, hnB, cF, cB, whh1fb, whh1bb, xwf, xwb,
        l1seqf, l1seqb, t);
  }

  // 5) hbar = tanh(l1 @ ws1^T)  (N padded 350->384; pad rows are zero)
  gemm128<false, true, false, true, false>
      <<<dim3(DAP / 128, MR / 128, 1), 256, 0, stream>>>(
      l1seqb, ws1b, nullptr, nullptr, hbar, H2, H2, H2, DAP, 0, 0, 0);

  // 6) alphas = hbar @ ws2^T   (K padded; ws2 pad cols are zero)
  gemm_bf16_Mx16<<<dim3(MR / 128), 256, 0, stream>>>(
      hbar, ws2b, alphas, DAP, DAP, DAP, R_);

  // 7) sent = attention-weighted sum of l1 (writes d_out + bf16 copy)
  sent_kernel<<<dim3(H2 / 256, B_), 256, 0, stream>>>(alphas, l1seqf, out, sentb);

  // 8) votes[b,r,:] = sent[b,r,:] @ caps_W[r]  (batched over r)
  gemm128<false, false, true, false, false>
      <<<dim3(H2 / 128, B_ / 128, R_), 256, 0, stream>>>(
      sentb, capsb, nullptr, votes, nullptr, H2,
      R_ * H2 /*lda*/, H2 /*ldw*/, R_ * H2 /*ldc*/,
      (long)H2 /*sA*/, (long)H2 * H2 /*sW*/, (long)H2 /*sC*/);

  // 9) dynamic routing -> class_logits appended after sent in d_out
  routing_kernel<<<dim3(B_), 256, 0, stream>>>(
      votes, out + (size_t)B_ * R_ * H2);
}
